// MixtureOfExperts_35871566856455
// MI455X (gfx1250) — matrix operand order repaired and checked by reading, not compiled, not run on hardware
//
#include <hip/hip_runtime.h>

typedef __bf16 bf16_t;
typedef __attribute__((ext_vector_type(16))) __bf16 v16bf;
typedef __attribute__((ext_vector_type(8)))  float  v8f;

#define BM 128
#define BN 128
#define BK 32
#define LDT 48   // LDS row stride in bf16 elements (96B: keeps b128 16B alignment)

__device__ __forceinline__ bf16_t f2bf(float f) { return (bf16_t)f; }

// Low 32 bits of a flat pointer to __shared__ = byte offset within LDS
// (LDS aperture base occupies addr[63:32] on CDNA5).
__device__ __forceinline__ uint32_t lds_off32(const void* p) {
  return (uint32_t)(uintptr_t)p;
}

// ---------------------------------------------------------------------------
// f32 -> bf16 elementwise conversion (for x_expert)
// ---------------------------------------------------------------------------
__global__ __launch_bounds__(256) void f32_to_bf16_kernel(
    const float* __restrict__ in, bf16_t* __restrict__ out, int n4) {
  int i = blockIdx.x * blockDim.x + threadIdx.x;
  if (i < n4) {
    float4 v = ((const float4*)in)[i];
    __align__(8) bf16_t o[4] = { f2bf(v.x), f2bf(v.y), f2bf(v.z), f2bf(v.w) };
    *(uint2*)(out + (size_t)i * 4) = *(const uint2*)o;
  }
}

// ---------------------------------------------------------------------------
// Tiled bf16 WMMA GEMM:  C[M,N] = epilogue( A[M,K](bf16) @ W[K,N](f32) + bias )
// MODE 0: relu, bf16 output (hidden layers: gating hidden g, expert hidden h)
// MODE 1: gate-weighted f32 accumulate into output (expert second layer)
// Block = 256 threads = 8 waves (2 in M x 4 in N); each wave owns 64x32 of C.
// 2-deep pipeline: A tile via GLOBAL_LOAD_ASYNC_TO_LDS (ASYNCcnt), W tile via
// f32 loads + v_cvt_pk_bf16 + transposed ds_store; one barrier per K-step.
// ---------------------------------------------------------------------------
template <int MODE>
__global__ __launch_bounds__(256) void moe_gemm_kernel(
    const bf16_t* __restrict__ A, int lda,
    const float* __restrict__ W, int ldw,
    const float* __restrict__ bias,
    void* __restrict__ Out, int ldo, int K,
    const float* __restrict__ gate, int expert, int accumulate) {
  __shared__ __align__(16) bf16_t lsA[2][BM * LDT];
  __shared__ __align__(16) bf16_t lsB[2][BN * LDT];

  const int tid   = threadIdx.x;
  const int lane  = tid & 31;
  const int wave  = tid >> 5;
  const int waveM = wave & 1;   // 2 waves along M
  const int waveN = wave >> 1;  // 4 waves along N
  const int row0  = blockIdx.y * BM;
  const int col0  = blockIdx.x * BN;

  v8f acc[4][2] = {};

  // A-tile async copy: each thread moves 16 contiguous bf16 (two b128 DMAs)
  const int arow = tid >> 1;
  const int acol = (tid & 1) * 16;
  // W-tile fill: each thread loads 4 rows x float4 (16 f32), converts+transposes
  const int kb = (tid >> 5) * 4;   // 0..28
  const int nb = (tid & 31) * 4;   // 0..124

  const uint32_t ldsA_base[2] = { lds_off32(&lsA[0][0]), lds_off32(&lsA[1][0]) };
  const uint32_t a_lds_rel = (uint32_t)(arow * LDT + acol) * 2u;

  auto async_fill_A = [&](int k0, int buf) {
    uint32_t loff = ldsA_base[buf] + a_lds_rel;
    const bf16_t* gp = A + (size_t)(row0 + arow) * lda + k0 + acol;
    asm volatile("global_load_async_to_lds_b128 %0, %1, off"
                 :: "v"(loff), "v"(gp) : "memory");
    asm volatile("global_load_async_to_lds_b128 %0, %1, off offset:16"
                 :: "v"(loff), "v"(gp) : "memory");
  };

  float4 wv[4];
  auto load_W = [&](int k0) {
#pragma unroll
    for (int i = 0; i < 4; ++i)
      wv[i] = *(const float4*)(W + (size_t)(k0 + kb + i) * ldw + col0 + nb);
  };

  // ---- pipeline prologue: stage tile 0 ----
  async_fill_A(0, 0);
  load_W(0);

  int cur = 0;
  for (int k0 = 0; k0 < K; k0 += BK, cur ^= 1) {
    bf16_t* sB = &lsB[cur][0];
    const bf16_t* sA = &lsA[cur][0];

    // ---- publish W tile (cvt f32->bf16, transposed [n][k] in LDS) ----
#pragma unroll
    for (int j = 0; j < 4; ++j) {
      __align__(8) bf16_t pk[4];
#pragma unroll
      for (int i = 0; i < 4; ++i) pk[i] = f2bf(((const float*)&wv[i])[j]);
      *(uint2*)&sB[(nb + j) * LDT + kb] = *(const uint2*)pk;
    }

    // own-wave async stores into lsA[cur] have landed; then publish to all
    asm volatile("s_wait_asynccnt 0" ::: "memory");
    __syncthreads();

    // ---- issue next tile's data movement (overlaps with WMMA below) ----
    if (k0 + BK < K) {
      async_fill_A(k0 + BK, cur ^ 1);
      load_W(k0 + BK);
#pragma unroll
      for (int i = 0; i < 4; ++i)
        __builtin_prefetch(W + (size_t)(k0 + 2 * BK + kb + i) * ldw + col0 + nb, 0, 1);
    }

    // ---- fragments per ISA 16-bit layouts ----
    union { v16bf v; uint4 u[2]; } afrag[4], bfrag[2];
    const int m15   = lane & 15;
    const int khalf = (lane >> 4) * 8;    // A: K groups {0..7,16..23} / {8..15,24..31}
    const int kbase = (lane >> 4) * 8;    // B: K groups {0..7,16..23} / {8..15,24..31}
#pragma unroll
    for (int ms = 0; ms < 4; ++ms) {
      int r = waveM * 64 + ms * 16 + m15;
      afrag[ms].u[0] = *(const uint4*)&sA[r * LDT + khalf];
      afrag[ms].u[1] = *(const uint4*)&sA[r * LDT + 16 + khalf];
    }
#pragma unroll
    for (int ns = 0; ns < 2; ++ns) {
      int c = waveN * 32 + ns * 16 + m15;
      bfrag[ns].u[0] = *(const uint4*)&sB[c * LDT + kbase];
      bfrag[ns].u[1] = *(const uint4*)&sB[c * LDT + kbase + 16];
    }
#pragma unroll
    for (int ms = 0; ms < 4; ++ms)
#pragma unroll
      for (int ns = 0; ns < 2; ++ns)
        acc[ms][ns] = __builtin_amdgcn_wmma_f32_16x16x32_bf16(
            false, afrag[ms].v, false, bfrag[ns].v,
            (short)0, acc[ms][ns], false, false);
  }

  // ---- epilogue (C layout: VGPR d -> M = d + 8*(lane>=16); N = lane&15) ----
#pragma unroll
  for (int ms = 0; ms < 4; ++ms) {
#pragma unroll
    for (int ns = 0; ns < 2; ++ns) {
      int cg  = col0 + waveN * 32 + ns * 16 + (lane & 15);
      float bv = bias[cg];
#pragma unroll
      for (int d = 0; d < 8; ++d) {
        int rg = row0 + waveM * 64 + ms * 16 + (lane >> 4) * 8 + d;
        float v = acc[ms][ns][d] + bv;
        if (MODE == 0) {
          v = v > 0.f ? v : 0.f;
          ((bf16_t*)Out)[(size_t)rg * ldo + cg] = f2bf(v);
        } else {
          float s = gate[(size_t)rg * 8 + expert];
          float* op = (float*)Out + (size_t)rg * ldo + cg;
          float r = s * v;
          if (accumulate) r += *op;
          *op = r;
        }
      }
    }
  }
}

// ---------------------------------------------------------------------------
// gate = softmax(g @ Wg2 + bg2): one wave32 per token, shfl_xor reduction
// ---------------------------------------------------------------------------
__global__ __launch_bounds__(256) void moe_gate_softmax_kernel(
    const bf16_t* __restrict__ g, const float* __restrict__ Wg2,
    const float* __restrict__ bg2, float* __restrict__ gate, int K) {
  const int lane  = threadIdx.x;  // 0..31
  const int token = blockIdx.x * blockDim.y + threadIdx.y;
  const bf16_t* grow = g + (size_t)token * K;
  float acc[8] = {};
  for (int k = lane; k < K; k += 32) {
    float gv = (float)grow[k];
    const float4* w = (const float4*)(Wg2 + (size_t)k * 8);
    float4 w0 = w[0], w1 = w[1];
    acc[0] += gv * w0.x; acc[1] += gv * w0.y;
    acc[2] += gv * w0.z; acc[3] += gv * w0.w;
    acc[4] += gv * w1.x; acc[5] += gv * w1.y;
    acc[6] += gv * w1.z; acc[7] += gv * w1.w;
  }
#pragma unroll
  for (int off = 16; off > 0; off >>= 1)
#pragma unroll
    for (int e = 0; e < 8; ++e) acc[e] += __shfl_xor(acc[e], off, 32);
  if (lane == 0) {
    float z[8], m = -1e30f, s = 0.f;
#pragma unroll
    for (int e = 0; e < 8; ++e) { z[e] = acc[e] + bg2[e]; m = fmaxf(m, z[e]); }
#pragma unroll
    for (int e = 0; e < 8; ++e) { z[e] = __expf(z[e] - m); s += z[e]; }
    float inv = 1.0f / s;
#pragma unroll
    for (int e = 0; e < 8; ++e) gate[(size_t)token * 8 + e] = z[e] * inv;
  }
}

// ---------------------------------------------------------------------------
extern "C" void kernel_launch(void* const* d_in, const int* in_sizes, int n_in,
                              void* d_out, int out_size, void* d_ws, size_t ws_size,
                              hipStream_t stream) {
  constexpr int B = 4096, D_IN = 1024, D_HID = 4096, D_OUT = 1024, E = 8;

  const float* x   = (const float*)d_in[0];
  const float* W1  = (const float*)d_in[1];
  const float* b1  = (const float*)d_in[2];
  const float* W2  = (const float*)d_in[3];
  const float* b2  = (const float*)d_in[4];
  const float* Wg1 = (const float*)d_in[5];
  const float* bg1 = (const float*)d_in[6];
  const float* Wg2 = (const float*)d_in[7];
  const float* bg2 = (const float*)d_in[8];
  float* out = (float*)d_out;

  // workspace layout (bytes):
  //   xb   : B*D_IN  bf16  =  8 MB
  //   g    : B*D_HID bf16  = 32 MB
  //   h    : B*D_HID bf16  = 32 MB   (reused per expert)
  //   gate : B*E     f32   = 128 KB
  char* ws = (char*)d_ws;
  bf16_t* xb   = (bf16_t*)(ws);
  bf16_t* gbuf = (bf16_t*)(ws + (size_t)B * D_IN * 2);
  bf16_t* hbuf = (bf16_t*)(ws + (size_t)B * D_IN * 2 + (size_t)B * D_HID * 2);
  float*  gate = (float*)(ws + (size_t)B * D_IN * 2 + 2 * (size_t)B * D_HID * 2);

  // 1) x -> bf16
  {
    int n4 = B * D_IN / 4;
    f32_to_bf16_kernel<<<dim3(n4 / 256), dim3(256), 0, stream>>>(x, xb, n4);
  }

  // 2) gating hidden: g = relu(xb @ Wg1 + bg1)   [4096 x 4096], K=1024
  moe_gemm_kernel<0><<<dim3(D_HID / BN, B / BM), dim3(256), 0, stream>>>(
      xb, D_IN, Wg1, D_HID, bg1, gbuf, D_HID, D_IN, nullptr, 0, 0);

  // 3) gate = softmax(g @ Wg2 + bg2)             [4096 x 8], K=4096
  moe_gate_softmax_kernel<<<dim3(B / 8), dim3(32, 8), 0, stream>>>(
      gbuf, Wg2, bg2, gate, D_HID);

  // 4) experts, serialized on the stream; out accumulates gate-weighted sums
  for (int e = 0; e < E; ++e) {
    const float* W1e = W1 + (size_t)e * D_IN * D_HID;
    const float* b1e = b1 + (size_t)e * D_HID;
    const float* W2e = W2 + (size_t)e * D_HID * D_OUT;
    const float* b2e = b2 + (size_t)e * D_OUT;

    // h = relu(xb @ W1[e] + b1[e])               [4096 x 4096], K=1024
    moe_gemm_kernel<0><<<dim3(D_HID / BN, B / BM), dim3(256), 0, stream>>>(
        xb, D_IN, W1e, D_HID, b1e, hbuf, D_HID, D_IN, nullptr, 0, 0);

    // out (+)= gate[:,e] * (h @ W2[e] + b2[e])   [4096 x 1024], K=4096
    moe_gemm_kernel<1><<<dim3(D_OUT / BN, B / BM), dim3(256), 0, stream>>>(
        hbuf, D_HID, W2e, D_OUT, b2e, out, D_OUT, D_HID, gate, e, e > 0 ? 1 : 0);
  }
}
